// SAGEConvolution_652835029486
// MI455X (gfx1250) — matrix operand order repaired and checked by reading, not compiled, not run on hardware
//
#include <hip/hip_runtime.h>
#include <hip/hip_bf16.h>

// ---------------- problem constants ----------------
#define F_IN   165
#define F_PAD  176          // 165 padded to multiple of 16
#define HID    256
#define NCLS   2
#define KTOT   352          // F_PAD * 2 (agg || x), 11 chunks of 32
#define KCHUNK 11
#define LDS_ROW 360         // A-tile LDS row stride in halfs (16B aligned, bank-spread)

typedef __attribute__((ext_vector_type(16))) __bf16        v16bf;
typedef __attribute__((ext_vector_type(8)))  float         v8f;
typedef __attribute__((ext_vector_type(4)))  unsigned int  v4u;

struct Frag32 { v4u lo, hi; };   // 32 bytes == one v16bf fragment

static __device__ __forceinline__ v16bf load_frag16(const unsigned short* p, int gap_halfs) {
    Frag32 f;
    f.lo = *(const v4u*)(p);
    f.hi = *(const v4u*)(p + gap_halfs);
    return __builtin_bit_cast(v16bf, f);
}

// ---------------- weight pre-pack: B-fragment order ----------------
// layout: [chunk 0..10][ntile 0..15][lane 0..31][j 0..15] contiguous halfs.
// element (chunk,ntile,lane,j) = combined-W[ K = chunk*32 + ((lane&16)>>1) + (j&7) + ((j&8)<<1) ][ N = ntile*16 + (lane&15) ]
// combined-W rows 0..164 = w1_l, rows 176..340 = w1_r, rest zero.
__global__ __launch_bounds__(256) void sage_pack_w1(const float* __restrict__ w1_l,
                                                    const float* __restrict__ w1_r,
                                                    __hip_bfloat16* __restrict__ wfrag) {
    int idx = blockIdx.x * 256 + threadIdx.x;           // 11*16*32*16 = 90112 total
    if (idx >= KCHUNK * 16 * 32 * 16) return;
    int j     = idx & 15;
    int lane  = (idx >> 4) & 31;
    int ntile = (idx >> 9) & 15;
    int chunk = idx >> 13;
    int K = chunk * 32 + ((lane & 16) >> 1) + (j & 7) + ((j & 8) << 1);
    int n = ntile * 16 + (lane & 15);
    float v = 0.0f;
    if (K < F_PAD)                 { if (K < F_IN)          v = w1_l[K * HID + n]; }
    else                           { int kk = K - F_PAD; if (kk < F_IN) v = w1_r[kk * HID + n]; }
    wfrag[idx] = __float2bfloat16(v);
}

// ---------------- x -> bf16 (padded) ----------------
__global__ __launch_bounds__(192) void sage_conv_x(const float* __restrict__ x,
                                                   __hip_bfloat16* __restrict__ xbf) {
    int i = blockIdx.x, t = threadIdx.x;
    if (t >= F_PAD) return;
    float v = (t < F_IN) ? x[(size_t)i * F_IN + t] : 0.0f;
    xbf[(size_t)i * F_PAD + t] = __float2bfloat16(v);
}

// ---------------- degree ----------------
__global__ __launch_bounds__(256) void sage_deg(const long long* __restrict__ ei,
                                                float* __restrict__ deg, int E) {
    int e = blockIdx.x * 256 + threadIdx.x;
    if (e < E) atomicAdd(&deg[(int)ei[(size_t)E + e]], 1.0f);
}

// ---------------- scatter layer 1: agg1[dst] += x[src] ----------------
__global__ __launch_bounds__(192) void sage_scatter1(const long long* __restrict__ ei,
                                                     const float* __restrict__ x,
                                                     float* __restrict__ agg1, int E) {
    int e = blockIdx.x, t = threadIdx.x;
    if (t >= F_IN) return;
    int src = (int)ei[e];
    int dst = (int)ei[(size_t)E + e];
    atomicAdd(&agg1[(size_t)dst * F_PAD + t], x[(size_t)src * F_IN + t]);
}

// ---------------- normalize + bf16 ----------------
__global__ __launch_bounds__(192) void sage_norm1(const float* __restrict__ agg1,
                                                  const float* __restrict__ deg,
                                                  __hip_bfloat16* __restrict__ aggbf) {
    int i = blockIdx.x, t = threadIdx.x;
    if (t >= F_PAD) return;
    float d = deg[i]; d = d > 1.0f ? d : 1.0f;
    float v = (t < F_IN) ? agg1[(size_t)i * F_PAD + t] / d : 0.0f;
    aggbf[(size_t)i * F_PAD + t] = __float2bfloat16(v);
}

// ---------------- layer-1 fused WMMA GEMM: h = relu([agg|x] @ Wcomb + b1) ----------------
// block = 256 threads (8 wave32); block tile = 16 rows x 256 cols; wave tile = 16x32 (2 WMMAs/chunk)
__global__ __launch_bounds__(256) void sage_gemm1(const __hip_bfloat16* __restrict__ aggbf,
                                                  const __hip_bfloat16* __restrict__ xbf,
                                                  const __hip_bfloat16* __restrict__ wfrag,
                                                  const float* __restrict__ bias1,
                                                  __hip_bfloat16* __restrict__ hbf, int N) {
    __shared__ unsigned short At[16 * LDS_ROW];     // 16 rows x 352 halfs (+pad)

    int rowbase = blockIdx.x * 16;
    // stage A: row = [aggbf(176 halfs) | xbf(176 halfs)] as dwords (88+88 per row)
    {
        const unsigned int* aggd = (const unsigned int*)aggbf;   // row stride 88 dwords
        const unsigned int* xd   = (const unsigned int*)xbf;
        unsigned int* Ad = (unsigned int*)At;                    // row stride LDS_ROW/2 = 180 dwords
        for (int t = threadIdx.x; t < 16 * (KTOT / 2); t += 256) {
            int r = t / (KTOT / 2);
            int c = t - r * (KTOT / 2);
            int grow = rowbase + r;
            unsigned int v = 0u;
            if (grow < N)
                v = (c < F_PAD / 2) ? aggd[(size_t)grow * (F_PAD / 2) + c]
                                    : xd[(size_t)grow * (F_PAD / 2) + (c - F_PAD / 2)];
            Ad[r * (LDS_ROW / 2) + c] = v;
        }
    }
    __syncthreads();

    int wave = threadIdx.x >> 5;
    int lane = threadIdx.x & 31;
    int colbase = wave * 32;
    int nt0 = colbase >> 4;          // ntile indices nt0, nt0+1

    int arow = lane & 15;
    int asel = (lane & 16) ? 8 : 0;  // A-fragment half-wave K offset

    v8f acc0 = {}; v8f acc1 = {};

    const unsigned short* wbase = (const unsigned short*)wfrag;
    for (int chunk = 0; chunk < KCHUNK; ++chunk) {
        const unsigned short* ap = &At[arow * LDS_ROW + chunk * 32 + asel];
        v16bf a = load_frag16(ap, 16);   // halfs [0..7] and [16..23]

        const unsigned short* bp0 = wbase + (((size_t)(chunk * 16 + nt0)     * 32 + lane) * 16);
        const unsigned short* bp1 = wbase + (((size_t)(chunk * 16 + nt0 + 1) * 32 + lane) * 16);
        v16bf b0 = load_frag16(bp0, 8);  // contiguous 32B block
        v16bf b1 = load_frag16(bp1, 8);

        acc0 = __builtin_amdgcn_wmma_f32_16x16x32_bf16(false, a, false, b0, (short)0, acc0, false, false);
        acc1 = __builtin_amdgcn_wmma_f32_16x16x32_bf16(false, a, false, b1, (short)0, acc1, false, false);
    }

    // C/D layout: VGPR r -> M = r (lanes 0-15) or r+8 (lanes 16-31); N = colbase + (lane&15)
    int n0   = colbase + (lane & 15);
    int mofs = (lane & 16) ? 8 : 0;
    float bb0 = bias1[n0], bb1 = bias1[n0 + 16];
    for (int r = 0; r < 8; ++r) {
        int m = rowbase + mofs + r;
        if (m < N) {
            float v0 = acc0[r] + bb0; v0 = v0 > 0.0f ? v0 : 0.0f;
            float v1 = acc1[r] + bb1; v1 = v1 > 0.0f ? v1 : 0.0f;
            hbf[(size_t)m * HID + n0]      = __float2bfloat16(v0);
            hbf[(size_t)m * HID + n0 + 16] = __float2bfloat16(v1);
        }
    }
}

// ---------------- scatter layer 2: agg2[dst] += h[src] ----------------
__global__ __launch_bounds__(HID) void sage_scatter2(const long long* __restrict__ ei,
                                                     const __hip_bfloat16* __restrict__ hbf,
                                                     float* __restrict__ agg2, int E) {
    int e = blockIdx.x, t = threadIdx.x;
    int src = (int)ei[e];
    int dst = (int)ei[(size_t)E + e];
    atomicAdd(&agg2[(size_t)dst * HID + t], __bfloat162float(hbf[(size_t)src * HID + t]));
}

__global__ __launch_bounds__(HID) void sage_norm2(const float* __restrict__ agg2,
                                                  const float* __restrict__ deg,
                                                  __hip_bfloat16* __restrict__ agg2bf) {
    int i = blockIdx.x, t = threadIdx.x;
    float d = deg[i]; d = d > 1.0f ? d : 1.0f;
    agg2bf[(size_t)i * HID + t] = __float2bfloat16(agg2[(size_t)i * HID + t] / d);
}

// ---------------- layer 2: logits = agg2 @ w2_l + h @ w2_r + b2 (N_CLASSES=2, VALU) ----------------
__global__ __launch_bounds__(256) void sage_out(const __hip_bfloat16* __restrict__ agg2bf,
                                                const __hip_bfloat16* __restrict__ hbf,
                                                const float* __restrict__ w2_l,
                                                const float* __restrict__ w2_r,
                                                const float* __restrict__ b2,
                                                float* __restrict__ out, int N) {
    __shared__ float wl[HID * NCLS];
    __shared__ float wr[HID * NCLS];
    __shared__ float bb[NCLS];
    for (int t = threadIdx.x; t < HID * NCLS; t += 256) { wl[t] = w2_l[t]; wr[t] = w2_r[t]; }
    if (threadIdx.x < NCLS) bb[threadIdx.x] = b2[threadIdx.x];
    __syncthreads();

    int i = blockIdx.x * 256 + threadIdx.x;
    if (i >= N) return;
    float s0 = bb[0], s1 = bb[1];
    const __hip_bfloat16* ap = agg2bf + (size_t)i * HID;
    const __hip_bfloat16* hp = hbf    + (size_t)i * HID;
    for (int k = 0; k < HID; ++k) {
        float a = __bfloat162float(ap[k]);
        float h = __bfloat162float(hp[k]);
        s0 += a * wl[k * 2 + 0] + h * wr[k * 2 + 0];
        s1 += a * wl[k * 2 + 1] + h * wr[k * 2 + 1];
    }
    out[(size_t)i * 2 + 0] = s0;
    out[(size_t)i * 2 + 1] = s1;
}

// ---------------- launch ----------------
extern "C" void kernel_launch(void* const* d_in, const int* in_sizes, int n_in,
                              void* d_out, int out_size, void* d_ws, size_t ws_size,
                              hipStream_t stream) {
    const float*     x    = (const float*)d_in[0];
    const long long* ei   = (const long long*)d_in[1];
    const float*     w1_l = (const float*)d_in[2];
    const float*     w1_r = (const float*)d_in[3];
    const float*     b1   = (const float*)d_in[4];
    const float*     w2_l = (const float*)d_in[5];
    const float*     w2_r = (const float*)d_in[6];
    const float*     b2   = (const float*)d_in[7];

    const int N = in_sizes[0] / F_IN;     // 200000
    const int E = in_sizes[1] / 2;        // 500000

    // workspace carve-out (256B aligned slabs)
    char*  ws  = (char*)d_ws;
    size_t off = 0;
    auto carve = [&](size_t bytes) -> void* {
        void* p = ws + off;
        off += (bytes + 255) & ~(size_t)255;
        return p;
    };
    float*          deg    = (float*)carve((size_t)N * sizeof(float));
    float*          agg1   = (float*)carve((size_t)N * F_PAD * sizeof(float));
    __hip_bfloat16* aggbf  = (__hip_bfloat16*)carve((size_t)N * F_PAD * 2);
    __hip_bfloat16* xbf    = (__hip_bfloat16*)carve((size_t)N * F_PAD * 2);
    __hip_bfloat16* wfrag  = (__hip_bfloat16*)carve((size_t)KTOT * HID * 2);
    __hip_bfloat16* hbf    = (__hip_bfloat16*)carve((size_t)N * HID * 2);
    float*          agg2   = (float*)carve((size_t)N * HID * sizeof(float));
    __hip_bfloat16* agg2bf = (__hip_bfloat16*)carve((size_t)N * HID * 2);

    // zero the atomic accumulators every call (graph-capture-safe memset nodes)
    hipMemsetAsync(deg,  0, (size_t)N * sizeof(float), stream);
    hipMemsetAsync(agg1, 0, (size_t)N * F_PAD * sizeof(float), stream);
    hipMemsetAsync(agg2, 0, (size_t)N * HID * sizeof(float), stream);

    // prep
    sage_pack_w1<<<(KCHUNK * 16 * 32 * 16 + 255) / 256, 256, 0, stream>>>(w1_l, w1_r, wfrag);
    sage_conv_x <<<N, 192, 0, stream>>>(x, xbf);
    sage_deg    <<<(E + 255) / 256, 256, 0, stream>>>(ei, deg, E);

    // layer 1
    sage_scatter1<<<E, 192, 0, stream>>>(ei, x, agg1, E);
    sage_norm1   <<<N, 192, 0, stream>>>(agg1, deg, aggbf);
    sage_gemm1   <<<(N + 15) / 16, 256, 0, stream>>>(aggbf, xbf, wfrag, b1, hbf, N);

    // layer 2
    sage_scatter2<<<E, HID, 0, stream>>>(ei, hbf, agg2, E);
    sage_norm2   <<<N, HID, 0, stream>>>(agg2, deg, agg2bf);
    sage_out     <<<(N + 255) / 256, 256, 0, stream>>>(agg2bf, hbf, w2_l, w2_r, b2, (float*)d_out, N);

    // tuple part 2: edge_index passthrough (int64 bytes appended after N*2 f32 logits)
    hipMemcpyAsync((char*)d_out + (size_t)N * NCLS * sizeof(float), ei,
                   (size_t)2 * E * sizeof(long long), hipMemcpyDeviceToDevice, stream);
}
